// SemiGCN_47820165873816
// MI455X (gfx1250) — matrix operand order repaired and checked by reading, not compiled, not run on hardware
//
#include <hip/hip_runtime.h>

typedef __attribute__((ext_vector_type(16))) _Float16 v16h;
typedef __attribute__((ext_vector_type(8)))  _Float16 v8h;
typedef __attribute__((ext_vector_type(8)))  float    v8f;

#define BN_INV 0.999995000037499687f   // 1/sqrt(1+1e-5)

// flags
#define F_BIAS    1
#define F_BNRELU  2
#define F_OUT_H   4
#define F_OUT_F   8
#define F_OUT_MAX 16

// ---------------------------------------------------------------------------
// Generic f16 WMMA GEMM: Y[M,O] = act(A[M,K] @ W[O,K]^T + bias) with fused
// BN(eval)+ReLU and optional per-batch channel max reduction (atomic).
// One wave per 16x16 output tile, K-loop in steps of 32 via v_wmma_f32_16x16x32_f16.
// ---------------------------------------------------------------------------
__global__ __launch_bounds__(256) void gemm_wmma_f16(
    const _Float16* __restrict__ A, const _Float16* __restrict__ W,
    const float* __restrict__ bias, const float* __restrict__ gamma,
    const float* __restrict__ beta,
    _Float16* __restrict__ Yh, float* __restrict__ Yf,
    float* __restrict__ Ymax,
    int M, int K, int O, int NperB, int flags)
{
    const int lane   = threadIdx.x & 31;
    const int waveId = blockIdx.x * (blockDim.x >> 5) + (threadIdx.x >> 5);
    const int tilesM = M >> 4;
    const int tilesO = O >> 4;
    if (waveId >= tilesM * tilesO) return;
    const int tm = waveId % tilesM;
    const int tn = waveId / tilesM;

    const int arow = tm * 16 + (lane & 15);
    const int bcol = tn * 16 + (lane & 15);
    const int a_kb = (lane >> 4) * 8;    // A: rows in lanes, K split 0..7 / 16..23 (+8 for hi lanes)
    const int b_kb = (lane >> 4) * 8;    // B: cols in lanes, K split 0..7 / 16..23 (+8 for hi lanes)

    const _Float16* __restrict__ ap = A + (size_t)arow * K;
    const _Float16* __restrict__ wp = W + (size_t)bcol * K;

    union AB { v16h v; v8h h[2]; };
    v8f acc = {};
    for (int kk = 0; kk < K; kk += 32) {
        AB a, b;
        a.h[0] = *(const v8h*)(ap + kk + a_kb);
        a.h[1] = *(const v8h*)(ap + kk + a_kb + 16);
        b.h[0] = *(const v8h*)(wp + kk + b_kb);
        b.h[1] = *(const v8h*)(wp + kk + b_kb + 16);
        acc = __builtin_amdgcn_wmma_f32_16x16x32_f16(
                  false, a.v, false, b.v, (short)0, acc, false, false);
    }

    // epilogue: lane holds 8 rows of one column
    const int col   = tn * 16 + (lane & 15);
    const int rbase = tm * 16 + ((lane >> 4) << 3);
    float bs = (flags & F_BIAS)   ? bias[col]          : 0.f;
    float sc = (flags & F_BNRELU) ? gamma[col] * BN_INV : 1.f;
    float sh = (flags & F_BNRELU) ? beta[col]          : 0.f;

    float vmax = 0.f;
    #pragma unroll
    for (int r = 0; r < 8; ++r) {
        float v = acc[r] + bs;
        if (flags & F_BNRELU) v = fmaxf(v * sc + sh, 0.f);
        const size_t idx = (size_t)(rbase + r) * O + col;
        if (flags & F_OUT_H) Yh[idx] = (_Float16)v;
        if (flags & F_OUT_F) Yf[idx] = v;
        vmax = fmaxf(vmax, v);
    }
    if (flags & F_OUT_MAX) {
        const int b = rbase / NperB;   // 16-row tile never straddles a batch (NperB=1024)
        atomicMax((unsigned int*)(Ymax + (size_t)b * O + col), __float_as_uint(vmax));
    }
}

// ---------------------------------------------------------------------------
// conv 3->64 with optional per-batch 3x3 transform on the input point,
// bias (optional) + BN + ReLU, f16 output [B*N, 64].  x layout: [B,3,N].
// ---------------------------------------------------------------------------
__global__ __launch_bounds__(64) void conv3_bn_relu_k(
    const float* __restrict__ x, const float* __restrict__ w,
    const float* __restrict__ bias, const float* __restrict__ g,
    const float* __restrict__ bt, const float* __restrict__ trans,
    _Float16* __restrict__ Y, int N)
{
    const int m = blockIdx.x;           // point index in [0, B*N)
    const int o = threadIdx.x;          // output channel
    const int b = m / N, n = m % N;
    const float* xb = x + (size_t)b * 3 * N + n;
    float p0 = xb[0], p1 = xb[N], p2 = xb[2 * N];
    if (trans) {
        const float* T = trans + b * 9;   // out[d] = sum_c x[c] * T[c*3+d]
        float q0 = p0 * T[0] + p1 * T[3] + p2 * T[6];
        float q1 = p0 * T[1] + p1 * T[4] + p2 * T[7];
        float q2 = p0 * T[2] + p1 * T[5] + p2 * T[8];
        p0 = q0; p1 = q1; p2 = q2;
    }
    float v = w[o * 3] * p0 + w[o * 3 + 1] * p1 + w[o * 3 + 2] * p2;
    if (bias) v += bias[o];
    v = fmaxf(g[o] * BN_INV * v + bt[o], 0.f);
    Y[(size_t)m * 64 + o] = (_Float16)v;
}

// small f32 fc: out[M,O] = in[M,K] @ w[O,K]^T + bias (+ I3 flattened if addIden)
__global__ void fc_small_k(const float* __restrict__ in, const float* __restrict__ w,
                           const float* __restrict__ bias, float* __restrict__ out,
                           int M, int K, int O, int addIden)
{
    const int t = blockIdx.x * blockDim.x + threadIdx.x;
    if (t >= M * O) return;
    const int m = t / O, o = t % O;
    const float* a  = in + (size_t)m * K;
    const float* ww = w  + (size_t)o * K;
    float s = bias[o];
    for (int k = 0; k < K; ++k) s += a[k] * ww[k];
    if (addIden && (o == 0 || o == 4 || o == 8)) s += 1.f;
    out[t] = s;
}

__global__ void f32_to_f16_k(const float* __restrict__ src, _Float16* __restrict__ dst, int n)
{
    for (int i = blockIdx.x * blockDim.x + threadIdx.x; i < n; i += gridDim.x * blockDim.x)
        dst[i] = (_Float16)src[i];
}

__global__ void zero_f32_k(float* __restrict__ p, int n)
{
    for (int i = blockIdx.x * blockDim.x + threadIdx.x; i < n; i += gridDim.x * blockDim.x)
        p[i] = 0.f;
}

// ---------------------------------------------------------------------------
// host side
// ---------------------------------------------------------------------------
enum {
    IN_X = 0,
    IN_SC1_W, IN_SC1_B, IN_SBN1_G, IN_SBN1_B,
    IN_SC2_W, IN_SC2_B, IN_SBN2_G, IN_SBN2_B,
    IN_SC3_W, IN_SC3_B, IN_SBN3_G, IN_SBN3_B,
    IN_SFC1_W, IN_SFC1_B, IN_SBN4_G, IN_SBN4_B,
    IN_SFC2_W, IN_SFC2_B, IN_SBN5_G, IN_SBN5_B,
    IN_SFC3_W, IN_SFC3_B,
    IN_GC1_W, IN_BN1_G, IN_BN1_B,
    IN_GC2_W, IN_BN2_G, IN_BN2_B,
    IN_GC3_W, IN_BN3_G, IN_BN3_B,
    IN_GC4_W, IN_BN4_G, IN_BN4_B,
    IN_GC5_W, IN_BN5_G, IN_BN5_B,
    IN_FC1_W, IN_FC1_B, IN_BN6_G, IN_BN6_B,
    IN_FC2_W, IN_FC2_B, IN_BN7_G, IN_BN7_B,
    IN_FC3_W, IN_FC3_B
};

static const int Bsz = 32, Npt = 1024, Mpt = Bsz * Npt, EMB = 1024;

extern "C" void kernel_launch(void* const* d_in, const int* in_sizes, int n_in,
                              void* d_out, int out_size, void* d_ws, size_t ws_size,
                              hipStream_t stream)
{
    const float* F(int i);
    auto f = [&](int i) { return (const float*)d_in[i]; };

    char* ws = (char*)d_ws;
    auto alloc = [&](size_t bytes) {
        char* p = ws;
        ws += (bytes + 255) & ~(size_t)255;
        return p;
    };

    // activations
    _Float16* bufA  = (_Float16*)alloc((size_t)Mpt * 64  * 2); // h1 / g1 / g3
    _Float16* bufB  = (_Float16*)alloc((size_t)Mpt * 128 * 2); // h2 / g4
    _Float16* bufC  = (_Float16*)alloc((size_t)Mpt * 64  * 2); // g2
    float*    max1  = (float*)   alloc((size_t)Bsz * EMB * 4); // STN pooled
    float*    max2  = (float*)   alloc((size_t)Bsz * EMB * 4); // trunk pooled
    _Float16* maxh  = (_Float16*)alloc((size_t)Bsz * EMB * 2);
    _Float16* s1h   = (_Float16*)alloc((size_t)Bsz * 512 * 2);
    float*    s2f   = (float*)   alloc((size_t)Bsz * 256 * 4);
    float*    trans = (float*)   alloc((size_t)Bsz * 9   * 4);
    _Float16* c1h   = (_Float16*)alloc((size_t)Bsz * 512 * 2);
    float*    c2f   = (float*)   alloc((size_t)Bsz * 256 * 4);

    // f16 weight copies (W stays [O,K] row-major; B-fragment loads want exactly that)
    struct WCvt { int idx; int n; _Float16* h; };
    WCvt wc[] = {
        { IN_SC2_W,  128 * 64,   nullptr }, { IN_SC3_W,  1024 * 128, nullptr },
        { IN_SFC1_W, 512 * 1024, nullptr }, { IN_SFC2_W, 256 * 512,  nullptr },
        { IN_GC2_W,  64 * 64,    nullptr }, { IN_GC3_W,  64 * 64,    nullptr },
        { IN_GC4_W,  128 * 64,   nullptr }, { IN_GC5_W,  1024 * 128, nullptr },
        { IN_FC1_W,  512 * 1024, nullptr }, { IN_FC2_W,  256 * 512,  nullptr },
    };
    for (auto& w : wc) w.h = (_Float16*)alloc((size_t)w.n * 2);
    for (auto& w : wc)
        f32_to_f16_k<<<(w.n + 255) / 256, 256, 0, stream>>>(f(w.idx), w.h, w.n);
    _Float16 *wh_sc2 = wc[0].h, *wh_sc3 = wc[1].h, *wh_sfc1 = wc[2].h, *wh_sfc2 = wc[3].h,
             *wh_gc2 = wc[4].h, *wh_gc3 = wc[5].h, *wh_gc4  = wc[6].h, *wh_gc5  = wc[7].h,
             *wh_fc1 = wc[8].h, *wh_fc2 = wc[9].h;

    // reset pooled-max accumulators every call (replay-safe, ReLU output >= 0)
    zero_f32_k<<<(Bsz * EMB + 255) / 256, 256, 0, stream>>>(max1, Bsz * EMB);
    zero_f32_k<<<(Bsz * EMB + 255) / 256, 256, 0, stream>>>(max2, Bsz * EMB);

    auto gemm = [&](const _Float16* A, const _Float16* W, const float* bias,
                    const float* g, const float* b, _Float16* Yh, float* Yf,
                    float* Ym, int M, int K, int O, int flags) {
        int tiles = (M >> 4) * (O >> 4);
        gemm_wmma_f16<<<(tiles + 7) / 8, 256, 0, stream>>>(
            A, W, bias, g, b, Yh, Yf, Ym, M, K, O, Npt, flags);
    };

    // ---- STN3d ----
    conv3_bn_relu_k<<<Mpt, 64, 0, stream>>>(f(IN_X), f(IN_SC1_W), f(IN_SC1_B),
                                            f(IN_SBN1_G), f(IN_SBN1_B), nullptr, bufA, Npt);
    gemm(bufA, wh_sc2, f(IN_SC2_B), f(IN_SBN2_G), f(IN_SBN2_B),
         bufB, nullptr, nullptr, Mpt, 64, 128, F_BIAS | F_BNRELU | F_OUT_H);
    gemm(bufB, wh_sc3, f(IN_SC3_B), f(IN_SBN3_G), f(IN_SBN3_B),
         nullptr, nullptr, max1, Mpt, 128, 1024, F_BIAS | F_BNRELU | F_OUT_MAX);
    f32_to_f16_k<<<(Bsz * EMB + 255) / 256, 256, 0, stream>>>(max1, maxh, Bsz * EMB);
    gemm(maxh, wh_sfc1, f(IN_SFC1_B), f(IN_SBN4_G), f(IN_SBN4_B),
         s1h, nullptr, nullptr, Bsz, 1024, 512, F_BIAS | F_BNRELU | F_OUT_H);
    gemm(s1h, wh_sfc2, f(IN_SFC2_B), f(IN_SBN5_G), f(IN_SBN5_B),
         nullptr, s2f, nullptr, Bsz, 512, 256, F_BIAS | F_BNRELU | F_OUT_F);
    fc_small_k<<<(Bsz * 9 + 255) / 256, 256, 0, stream>>>(
        s2f, f(IN_SFC3_W), f(IN_SFC3_B), trans, Bsz, 256, 9, 1);

    // ---- transform + trunk (knn/adjacency is dead code in the reference) ----
    conv3_bn_relu_k<<<Mpt, 64, 0, stream>>>(f(IN_X), f(IN_GC1_W), nullptr,
                                            f(IN_BN1_G), f(IN_BN1_B), trans, bufA, Npt);
    gemm(bufA, wh_gc2, nullptr, f(IN_BN2_G), f(IN_BN2_B),
         bufC, nullptr, nullptr, Mpt, 64, 64, F_BNRELU | F_OUT_H);
    gemm(bufC, wh_gc3, nullptr, f(IN_BN3_G), f(IN_BN3_B),
         bufA, nullptr, nullptr, Mpt, 64, 64, F_BNRELU | F_OUT_H);
    gemm(bufA, wh_gc4, nullptr, f(IN_BN4_G), f(IN_BN4_B),
         bufB, nullptr, nullptr, Mpt, 64, 128, F_BNRELU | F_OUT_H);
    gemm(bufB, wh_gc5, nullptr, f(IN_BN5_G), f(IN_BN5_B),
         nullptr, nullptr, max2, Mpt, 128, 1024, F_BNRELU | F_OUT_MAX);

    // ---- classifier head ----
    f32_to_f16_k<<<(Bsz * EMB + 255) / 256, 256, 0, stream>>>(max2, maxh, Bsz * EMB);
    gemm(maxh, wh_fc1, f(IN_FC1_B), f(IN_BN6_G), f(IN_BN6_B),
         c1h, nullptr, nullptr, Bsz, 1024, 512, F_BIAS | F_BNRELU | F_OUT_H);
    gemm(c1h, wh_fc2, f(IN_FC2_B), f(IN_BN7_G), f(IN_BN7_B),
         nullptr, c2f, nullptr, Bsz, 512, 256, F_BIAS | F_BNRELU | F_OUT_F);
    fc_small_k<<<(Bsz * 40 + 255) / 256, 256, 0, stream>>>(
        c2f, f(IN_FC3_W), f(IN_FC3_B), (float*)d_out, Bsz, 256, 40, 0);
}
